// TinyMoE_2027224563962
// MI455X (gfx1250) — compile-verified, matrix operand order checked by reading, not run on hardware
//
#include <hip/hip_runtime.h>

// ---------------- CDNA5 types ----------------
typedef __attribute__((ext_vector_type(16))) __bf16   v16bf;
typedef __attribute__((ext_vector_type(8)))  float    v8f;
typedef __attribute__((ext_vector_type(4)))  float    f32x4;
typedef __attribute__((ext_vector_type(4)))  unsigned u32x4;
typedef __attribute__((ext_vector_type(8)))  int      i32x8;
typedef __attribute__((ext_vector_type(4)))  int      i32x4;

union Frag { v16bf v; u32x4 q[2]; };   // one 16-bit WMMA A/B operand (32 B)

#define HDIM  1024
#define NEXP  8
#define PITCH 40      // GEMM LDS row pitch in halfwords: 80B rows, conflict-free b128 reads
#define TPIT  72      // transpose-kernel LDS tile pitch (144B rows, 16B aligned)

#ifndef __has_builtin
#define __has_builtin(x) 0
#endif
#if __has_builtin(__builtin_amdgcn_tensor_load_to_lds) && __has_builtin(__builtin_amdgcn_s_wait_tensorcnt)
#define HAVE_TDM 1
#else
#define HAVE_TDM 0
#endif

__device__ __forceinline__ unsigned short f2bf(float f) {
    union { float f; unsigned u; } v; v.f = f;
    unsigned r = v.u + 0x7FFFu + ((v.u >> 16) & 1u);   // RNE
    return (unsigned short)(r >> 16);
}
__device__ __forceinline__ unsigned pk2(float lo, float hi) {
    return (unsigned)f2bf(lo) | ((unsigned)f2bf(hi) << 16);
}

// ---------------- Kernel 1: router -------------------------------------------------------
__global__ __launch_bounds__(256) void moe_router(
    const float* __restrict__ x, const float* __restrict__ rw,
    int* __restrict__ cnt, int* __restrict__ list,
    float* __restrict__ wlist, float* __restrict__ bias, int N)
{
    __shared__ float rs[NEXP * HDIM];
    int tid = threadIdx.x;
    for (int i = tid * 4; i < NEXP * HDIM; i += 256 * 4)
        *(f32x4*)&rs[i] = *(const f32x4*)&rw[i];
    __syncthreads();

    int t = blockIdx.x * 256 + tid;
    if (t >= N) return;
    const float* xr = x + (size_t)t * HDIM;

    float acc[NEXP];
#pragma unroll
    for (int e = 0; e < NEXP; ++e) acc[e] = 0.f;
    for (int h = 0; h < HDIM; h += 4) {
        f32x4 xv = *(const f32x4*)(xr + h);
#pragma unroll
        for (int e = 0; e < NEXP; ++e) {
            f32x4 wv = *(const f32x4*)&rs[e * HDIM + h];
            acc[e] += xv.x * wv.x + xv.y * wv.y + xv.z * wv.z + xv.w * wv.w;
        }
    }
    float m = acc[0];
#pragma unroll
    for (int e = 1; e < NEXP; ++e) m = fmaxf(m, acc[e]);
    float s = 0.f, p[NEXP];
#pragma unroll
    for (int e = 0; e < NEXP; ++e) { p[e] = __expf(acc[e] - m); s += p[e]; }
    float inv = 1.f / s;
#pragma unroll
    for (int e = 0; e < NEXP; ++e) p[e] *= inv;
    int e0 = 0; float v0 = p[0];
#pragma unroll
    for (int e = 1; e < NEXP; ++e) if (p[e] > v0) { v0 = p[e]; e0 = e; }
    int e1 = (e0 == 0) ? 1 : 0; float v1 = p[e1];
#pragma unroll
    for (int e = 0; e < NEXP; ++e) if (e != e0 && p[e] > v1) { v1 = p[e]; e1 = e; }

    float wsum = v0 + v1 + 1e-6f;
    float w0 = v0 / wsum, w1 = v1 / wsum;
    bias[t] = w0 * v0 + w1 * v1;

    int p0 = atomicAdd(&cnt[e0], 1);
    list[e0 * N + p0] = t;  wlist[e0 * N + p0] = w0;
    int p1 = atomicAdd(&cnt[e1], 1);
    list[e1 * N + p1] = t;  wlist[e1 * N + p1] = w1;
}

// ---------------- Kernel 2a: x (fp32) -> xbf (bf16), same layout -------------------------
__global__ __launch_bounds__(256) void cvt_x(
    const float* __restrict__ x, unsigned short* __restrict__ xbf)
{
    size_t i = ((size_t)blockIdx.x * 256 + threadIdx.x) * 8;
    f32x4 a = *(const f32x4*)(x + i);
    f32x4 b = *(const f32x4*)(x + i + 4);
    u32x4 o = (u32x4){ pk2(a.x,a.y), pk2(a.z,a.w), pk2(b.x,b.y), pk2(b.z,b.w) };
    *(u32x4*)&xbf[i] = o;
}

// ---------------- Kernel 2b: W[e][k][n] (fp32) -> WbfT[e][n][k] (bf16, transposed) -------
__global__ __launch_bounds__(256) void cvt_wt(
    const float* __restrict__ W, unsigned short* __restrict__ wbt)
{
    __shared__ unsigned short tile[64 * TPIT];     // [n_local][k_local] bf16
    const int e  = blockIdx.z;
    const int k0 = blockIdx.x * 64;
    const int n0 = blockIdx.y * 64;
    const int tid = threadIdx.x;

    // read phase: rows k0+2p, k0+2p+1; 8 n's per thread; pack pairs across k
    {
        int p  = tid >> 3;            // 0..31 -> k-pair
        int nc = (tid & 7) * 8;       // 0..56
        const float* r0 = W + ((size_t)e * HDIM + (k0 + 2 * p)) * HDIM + n0 + nc;
        const float* r1 = r0 + HDIM;
        f32x4 a0 = *(const f32x4*)(r0);
        f32x4 a1 = *(const f32x4*)(r0 + 4);
        f32x4 b0 = *(const f32x4*)(r1);
        f32x4 b1 = *(const f32x4*)(r1 + 4);
        unsigned us[8] = { pk2(a0.x,b0.x), pk2(a0.y,b0.y), pk2(a0.z,b0.z), pk2(a0.w,b0.w),
                           pk2(a1.x,b1.x), pk2(a1.y,b1.y), pk2(a1.z,b1.z), pk2(a1.w,b1.w) };
#pragma unroll
        for (int j = 0; j < 8; ++j)
            *(unsigned*)&tile[(nc + j) * TPIT + 2 * p] = us[j];
    }
    __syncthreads();
    // write phase: contiguous 16-k chunks per thread, coalesced b128 stores
    {
        int nl = tid >> 2;            // 0..63
        int c  = (tid & 3) * 16;      // 0..48
        u32x4 q0 = *(u32x4*)&tile[nl * TPIT + c];
        u32x4 q1 = *(u32x4*)&tile[nl * TPIT + c + 8];
        unsigned short* dst = wbt + ((size_t)e * HDIM + n0 + nl) * HDIM + k0 + c;
        *(u32x4*)(dst)     = q0;
        *(u32x4*)(dst + 8) = q1;
    }
}

// ---------------- Kernel 3: out[n,:] = bias[n] -------------------------------------------
__global__ __launch_bounds__(256) void moe_init_out(
    const float* __restrict__ bias, float* __restrict__ out)
{
    int n = blockIdx.x;
    float b = bias[n];
    f32x4 v = { b, b, b, b };
    *(f32x4*)&out[(size_t)n * HDIM + threadIdx.x * 4] = v;
}

// ---------------- TDM descriptor: 128(n-rows) x 32(k) bf16 tile, stride H, pad to PITCH ---
#if HAVE_TDM
__device__ __forceinline__ void tdm_load_B(const unsigned short* gptr, unsigned lds_off) {
    unsigned long long ga = (unsigned long long)(uintptr_t)gptr;
    u32x4 g0;
    g0[0] = 1u;                                             // count=1, user mode, no gather
    g0[1] = lds_off;                                        // lds_addr (bytes)
    g0[2] = (unsigned)(ga & 0xFFFFFFFFu);                   // global_addr[31:0] (tile start)
    g0[3] = ((unsigned)(ga >> 32) & 0x01FFFFFFu) | 0x80000000u;  // addr[56:32] | type=2
    i32x8 g1;
    g1[0] = (int)((1u << 16) |       // data_size = 2 bytes (bf16)
                  (1u << 20) |       // pad_enable
                  (3u << 22) |       // pad_interval: 16 DWORDs (=64B row)
                  (3u << 25));       // pad_amount: 4 DWORDs (=16B -> 80B pitch = PITCH)
    g1[1] = (int)(1024u << 16);      // tensor_dim0 = 1024 (k)
    g1[2] = (int)(1024u << 16);      // tensor_dim0 hi=0 | tensor_dim1 = 1024 (n) low16
    g1[3] = (int)(32u << 16);        // tensor_dim1 hi=0 | tile_dim0 = 32 (k per step)
    g1[4] = 128;                     // tile_dim1 = 128 (n rows), tile_dim2 = 0
    g1[5] = 1024;                    // tensor_dim0_stride = 1024 elements
    g1[6] = 0;
    g1[7] = 0;
    i32x4 z = { 0, 0, 0, 0 };
#if __clang_major__ >= 23
    i32x8 z8 = { 0, 0, 0, 0, 0, 0, 0, 0 };
    __builtin_amdgcn_tensor_load_to_lds((u32x4)g0, (i32x8)g1, z, z, z8, 0);
#else
    __builtin_amdgcn_tensor_load_to_lds((u32x4)g0, (i32x8)g1, z, z, 0);
#endif
}
#endif

// ---------------- Kernel 4: grouped expert GEMM, BF16 WMMA, double-buffered --------------
__global__ __launch_bounds__(256) void moe_gemm(
    const unsigned short* __restrict__ xbf, const unsigned short* __restrict__ wbt,
    const int* __restrict__ cnt, const int* __restrict__ list,
    const float* __restrict__ wlist, float* __restrict__ out, int N)
{
    __shared__ unsigned short As[2][128 * PITCH];   // gathered token rows [row][k]
    __shared__ unsigned short Bs[2][128 * PITCH];   // WbfT rows          [n][k]
    __shared__ int   tk[128];
    __shared__ float wt[128];

    const int mtiles = N >> 7;
    const int e  = blockIdx.x / mtiles;
    const int m0 = (blockIdx.x % mtiles) << 7;
    const int count = cnt[e];
    if (m0 >= count) return;
    const int n0 = blockIdx.y << 7;

    const int tid = threadIdx.x;
    if (tid < 128) {
        int gm = m0 + tid;
        bool valid = gm < count;
        tk[tid] = valid ? list[e * N + gm] : 0;
        wt[tid] = valid ? wlist[e * N + gm] : 0.f;
    }
    __syncthreads();

    // wave mapping: 8 waves -> 2(M) x 4(N), each wave: 64x32 region = 4x2 C tiles
    const int lane  = tid & 31;
    const int wave  = tid >> 5;
    const int waveM = (wave & 1) * 64;
    const int waveN = (wave >> 1) * 32;
    const int half  = lane >> 4;
    const int lrow  = lane & 15;
    const int c0    = half * 8;

    v8f acc[4][2];
#pragma unroll
    for (int mi = 0; mi < 4; ++mi)
#pragma unroll
        for (int ni = 0; ni < 2; ++ni)
            acc[mi][ni] = (v8f){0.f,0.f,0.f,0.f,0.f,0.f,0.f,0.f};

    // A staging role: 2 threads per row, 32B each (already-bf16 copies, zero conversion)
    const int arow = tid >> 1;
    const int ah   = (tid & 1) * 16;
    const unsigned short* xrow = xbf + (size_t)tk[arow] * HDIM + ah;
    const unsigned short* bbase = wbt + ((size_t)e * HDIM + n0) * HDIM;   // tile (n0, k)
#if !HAVE_TDM
    const unsigned short* brow = bbase + (size_t)arow * HDIM + ah;
#endif

#define STAGE_A(buf, k0)                                                    \
    do {                                                                    \
        u32x4 v0 = *(const u32x4*)(xrow + (k0));                            \
        u32x4 v1 = *(const u32x4*)(xrow + (k0) + 8);                        \
        *(u32x4*)&As[buf][arow * PITCH + ah]     = v0;                      \
        *(u32x4*)&As[buf][arow * PITCH + ah + 8] = v1;                      \
    } while (0)
#if !HAVE_TDM
#define STAGE_B(buf, k0)                                                    \
    do {                                                                    \
        u32x4 v0 = *(const u32x4*)(brow + (k0));                            \
        u32x4 v1 = *(const u32x4*)(brow + (k0) + 8);                        \
        *(u32x4*)&Bs[buf][arow * PITCH + ah]     = v0;                      \
        *(u32x4*)&Bs[buf][arow * PITCH + ah + 8] = v1;                      \
    } while (0)
#endif

#if HAVE_TDM
    const unsigned ldsB[2] = { (unsigned)(uintptr_t)&Bs[0][0],
                               (unsigned)(uintptr_t)&Bs[1][0] };
#endif

    int buf = 0;
    STAGE_A(0, 0);
#if HAVE_TDM
    if (tid < 32) tdm_load_B(bbase, ldsB[0]);
#else
    STAGE_B(0, 0);
#endif

    for (int k0 = 0; k0 < HDIM; k0 += 32) {
#if HAVE_TDM
        if (tid < 32) __builtin_amdgcn_s_wait_tensorcnt(0);   // wave0's TDM done
#endif
        __syncthreads();                                      // staging for `buf` visible
        const int nxt = buf ^ 1;
        if (k0 + 32 < HDIM) {                                 // prefetch next K-step
            STAGE_A(nxt, k0 + 32);
#if HAVE_TDM
            if (tid < 32) tdm_load_B(bbase + k0 + 32, ldsB[nxt]);
#else
            STAGE_B(nxt, k0 + 32);
#endif
        }
        // ---- compute on `buf`: 12 ds_load_b128 + 8 WMMA ----
        const unsigned short* Ab = &As[buf][0];
        const unsigned short* Bb = &Bs[buf][0];
        Frag afr[4], bfr[2];
#pragma unroll
        for (int mi = 0; mi < 4; ++mi) {
            int r = waveM + mi * 16 + lrow;
            afr[mi].q[0] = *(const u32x4*)&Ab[r * PITCH + c0];
            afr[mi].q[1] = *(const u32x4*)&Ab[r * PITCH + c0 + 16];
        }
#pragma unroll
        for (int ni = 0; ni < 2; ++ni) {
            int r = waveN + ni * 16 + lrow;
            bfr[ni].q[0] = *(const u32x4*)&Bb[r * PITCH + c0];
            bfr[ni].q[1] = *(const u32x4*)&Bb[r * PITCH + c0 + 16];
        }
#pragma unroll
        for (int mi = 0; mi < 4; ++mi)
#pragma unroll
            for (int ni = 0; ni < 2; ++ni)
                acc[mi][ni] = __builtin_amdgcn_wmma_f32_16x16x32_bf16(
                    false, afr[mi].v, false, bfr[ni].v,
                    (short)0, acc[mi][ni], false, false);
        buf = nxt;
    }

    // ---- epilogue: C layout (VGPR r -> row r + 8*half, col = lane&15); atomic combine ----
#pragma unroll
    for (int mi = 0; mi < 4; ++mi) {
#pragma unroll
        for (int ni = 0; ni < 2; ++ni) {
            int gcol = n0 + waveN + ni * 16 + lrow;
#pragma unroll
            for (int vr = 0; vr < 8; ++vr) {
                int rl = waveM + mi * 16 + vr + 8 * half;
                float wv = wt[rl];
                int   t  = tk[rl];
                atomicAdd(out + (size_t)t * HDIM + gcol, wv * acc[mi][ni][vr]);
            }
        }
    }
#undef STAGE_A
#undef STAGE_B
}

// ---------------- launch -----------------------------------------------------------------
extern "C" void kernel_launch(void* const* d_in, const int* in_sizes, int n_in,
                              void* d_out, int out_size, void* d_ws, size_t ws_size,
                              hipStream_t stream) {
    const float* x  = (const float*)d_in[0];   // [N, H]
    const float* rw = (const float*)d_in[1];   // [E, H]
    const float* W  = (const float*)d_in[2];   // [E, H, H]
    float* out = (float*)d_out;

    const int N = in_sizes[0] / HDIM;          // 8192

    char* ws = (char*)d_ws;
    size_t off = 0;
    int*   cnt   = (int*)(ws + off);  off += 256;
    int*   list  = (int*)(ws + off);  off += (size_t)NEXP * N * 4;
    float* wlist = (float*)(ws + off); off += (size_t)NEXP * N * 4;
    float* bias  = (float*)(ws + off); off += (size_t)N * 4;
    off = (off + 255) & ~(size_t)255;
    unsigned short* xbf = (unsigned short*)(ws + off); off += (size_t)N * HDIM * 2;
    unsigned short* wbt = (unsigned short*)(ws + off); off += (size_t)NEXP * HDIM * HDIM * 2;

    hipMemsetAsync(cnt, 0, NEXP * sizeof(int), stream);
    moe_router<<<(N + 255) / 256, 256, 0, stream>>>(x, rw, cnt, list, wlist, bias, N);
    cvt_x<<<(int)(((size_t)N * HDIM) / 2048), 256, 0, stream>>>(x, xbf);
    cvt_wt<<<dim3(HDIM / 64, HDIM / 64, NEXP), 256, 0, stream>>>(W, wbt);
    moe_init_out<<<N, 256, 0, stream>>>(bias, out);
    moe_gemm<<<dim3(NEXP * (N / 128), HDIM / 128), 256, 0, stream>>>(xbf, wbt, cnt, list, wlist, out, N);
}
